// RTEmbedding_75402445848762
// MI455X (gfx1250) — matrix operand order repaired and checked by reading, not compiled, not run on hardware
//
#include <hip/hip_runtime.h>
#include <hip/hip_bf16.h>

typedef __bf16 bf16_t;
typedef __attribute__((ext_vector_type(16))) __bf16 v16bf;
typedef __attribute__((ext_vector_type(8)))  __bf16 v8bf;
typedef __attribute__((ext_vector_type(4)))  __bf16 v4bf;
typedef __attribute__((ext_vector_type(8)))  float  v8f;
typedef __attribute__((ext_vector_type(4)))  float  v4f;

#define N_U 20000
#define N_O 10000
#define CDIM 256
#define TXTD 1536
#define NNUM 3
#define NCAT 3
#define NCOLS 7
#define VOCAB 101
#define E_EDGES 10000
#define MAXP 16
#define NUM_NODES (N_U + N_O)
#define NTOK (NCOLS * NUM_NODES)   /* 210000 */

/* d_out layout (floats, reference tuple order) */
#define OFF_X    0ll
#define OFF_NODE ((long long)NTOK * CDIM)
#define OFF_COL  (OFF_NODE + NTOK)
#define OFF_TAB  (OFF_COL + NTOK)
#define OFF_F2P  (OFF_TAB + NTOK)
#define OFF_NN   (OFF_F2P + (long long)NTOK * MAXP)

/* ---------------- numeric + categorical tokens ---------------- */
__global__ void tok_numcat(const float* __restrict__ num, const int* __restrict__ cat,
                           const float* __restrict__ ctab, const float* __restrict__ nw,
                           const float* __restrict__ nb, const float* __restrict__ colw,
                           const float* __restrict__ temb, float* __restrict__ x,
                           int N, long long rowBase)
{
  int c = threadIdx.x;
  int j = blockIdx.x % (NNUM + NCAT);
  int n = blockIdx.x / (NNUM + NCAT);
  if (n >= N) return;
  float val;
  if (j < NNUM) {
    float v = num[n * NNUM + j];
    float t = v * nw[j * CDIM + c] + nb[j * CDIM + c];
    val = t / (1.0f + __expf(-t));           /* silu */
  } else {
    int k = j - NNUM;
    int idx = cat[n * NCAT + k];
    val = ctab[((size_t)k * VOCAB + idx) * CDIM + c];
  }
  val += colw[j * CDIM + c] + temb[c];
  x[(rowBase + (long long)j * N + n) * CDIM + c] = val;
}

/* ---------------- text-column GEMM: bf16 WMMA, f32 accumulate ---------------- */
#define KCH   128           /* K chunk staged in LDS   */
#define LDB   136           /* bf16 elems per LDS row (272B: aligned, bank-staggered) */
#define MTILE 128           /* rows per block (8 waves x 16) */

__global__ __launch_bounds__(256) void text_gemm_wmma(
    const float* __restrict__ A,      /* N x 1536 */
    const float* __restrict__ W,      /* 256 x 1536 (row-major, K contiguous) */
    const float* __restrict__ bias,   /* 256 */
    const float* __restrict__ colw,   /* 256 (col row 6)  */
    const float* __restrict__ temb,   /* 256 */
    float* __restrict__ xout,         /* pre-offset to text rows */
    int Nrows)
{
  __shared__ bf16_t Bs[128 * LDB];    /* ~34 KB of the 320 KB WGP LDS */
  const int tid  = threadIdx.x;
  const int wv   = tid >> 5;
  const int lane = tid & 31;
  const int kh   = lane >> 4;         /* wave half */
  const int l16  = lane & 15;
  const int mBase = blockIdx.x * MTILE + wv * 16;
  const int nBase = blockIdx.y * 128;

  v8f acc[8];
#pragma unroll
  for (int i = 0; i < 8; ++i)
#pragma unroll
    for (int q = 0; q < 8; ++q) acc[i][q] = 0.0f;

  int aRow = mBase + l16; if (aRow >= Nrows) aRow = Nrows - 1;   /* clamp; stores predicated */
  const float* aPtr = A + (size_t)aRow * TXTD;

  for (int kc = 0; kc < TXTD; kc += KCH) {
    __syncthreads();
    /* stage W chunk -> LDS bf16 (coalesced 16B loads, 8B LDS stores) */
#pragma unroll
    for (int it = 0; it < (128 * KCH / 4) / 256; ++it) {
      int linear = (it * 256 + tid) * 4;
      int n = linear / KCH;
      int k = linear % KCH;
      v4f w = *(const v4f*)(W + (size_t)(nBase + n) * TXTD + kc + k);
      v4bf b;
      b[0] = (bf16_t)w[0]; b[1] = (bf16_t)w[1];
      b[2] = (bf16_t)w[2]; b[3] = (bf16_t)w[3];
      *(v4bf*)(&Bs[n * LDB + k]) = b;
    }
    __syncthreads();

#pragma unroll
    for (int ks = 0; ks < KCH; ks += 32) {
      const int kb = kc + ks + kh * 8;
      /* A fragment: elems 0..7 = K+0..7, elems 8..15 = K+16..23 (per half) */
      v4f f0 = *(const v4f*)(aPtr + kb);
      v4f f1 = *(const v4f*)(aPtr + kb + 4);
      v4f f2 = *(const v4f*)(aPtr + kb + 16);
      v4f f3 = *(const v4f*)(aPtr + kb + 20);
      v16bf a;
      a[0]=(bf16_t)f0[0];  a[1]=(bf16_t)f0[1];  a[2]=(bf16_t)f0[2];  a[3]=(bf16_t)f0[3];
      a[4]=(bf16_t)f1[0];  a[5]=(bf16_t)f1[1];  a[6]=(bf16_t)f1[2];  a[7]=(bf16_t)f1[3];
      a[8]=(bf16_t)f2[0];  a[9]=(bf16_t)f2[1];  a[10]=(bf16_t)f2[2]; a[11]=(bf16_t)f2[3];
      a[12]=(bf16_t)f3[0]; a[13]=(bf16_t)f3[1]; a[14]=(bf16_t)f3[2]; a[15]=(bf16_t)f3[3];

      /* B fragments: lane = N col, elems 0..7 = K+0..7, elems 8..15 = K+16..23 (half*8 offset).
         Double-buffered so tile nt+1's ds_loads overlap tile nt's WMMA. */
      const bf16_t* brow = &Bs[l16 * LDB + ks + kh * 8];
      v16bf bfrag[2];
      {
        v8bf blo = *(const v8bf*)(brow);
        v8bf bhi = *(const v8bf*)(brow + 16);
        bfrag[0] = __builtin_shufflevector(blo, bhi,
                     0,1,2,3,4,5,6,7,8,9,10,11,12,13,14,15);
      }
#pragma unroll
      for (int nt = 0; nt < 8; ++nt) {
        if (nt < 7) {
          const bf16_t* bp = brow + (nt + 1) * (16 * LDB);
          v8bf blo = *(const v8bf*)(bp);
          v8bf bhi = *(const v8bf*)(bp + 16);
          bfrag[(nt + 1) & 1] = __builtin_shufflevector(blo, bhi,
                                  0,1,2,3,4,5,6,7,8,9,10,11,12,13,14,15);
        }
        acc[nt] = __builtin_amdgcn_wmma_f32_16x16x32_bf16(
            false, a, false, bfrag[nt & 1], (short)0, acc[nt], false, false);
      }
    }
  }

  /* epilogue: + tb + col + table_emb, write D (lane=N, vgpr=M) */
#pragma unroll
  for (int nt = 0; nt < 8; ++nt) {
    int c = nBase + nt * 16 + l16;
    float bb = bias[c] + colw[c] + temb[c];
#pragma unroll
    for (int v = 0; v < 8; ++v) {
      int r = mBase + kh * 8 + v;
      if (r < Nrows) xout[(size_t)r * CDIM + c] = acc[nt][v] + bb;
    }
  }
}

/* ---------------- index arrays + num_nodes ---------------- */
__global__ void idx_kernel(float* __restrict__ out) {
  long long i = (long long)blockIdx.x * blockDim.x + threadIdx.x;
  if (i >= NTOK) return;
  float node, col, tab;
  if (i < (long long)NCOLS * N_U) {
    node = (float)(int)(i % N_U);
    col  = (float)(int)(i / N_U);
    tab  = 0.0f;
  } else {
    long long ii = i - (long long)NCOLS * N_U;
    node = (float)(N_U + (int)(ii % N_O));
    col  = (float)(NCOLS + (int)(ii / N_O));
    tab  = 1.0f;
  }
  out[OFF_NODE + i] = node;
  out[OFF_COL + i]  = col;
  out[OFF_TAB + i]  = tab;
  if (i == 0) out[OFF_NN] = (float)NUM_NODES;
}

/* ---------------- f2p neighbor table ---------------- */
__global__ void f2p_init(float* __restrict__ out) {
  long long i = (long long)blockIdx.x * blockDim.x + threadIdx.x;
  if (i < (long long)NTOK * MAXP) out[OFF_F2P + i] = -1.0f;
}

__global__ void zero_cnt(int* __restrict__ cnt) {
  int i = blockIdx.x * blockDim.x + threadIdx.x;
  if (i < N_O) cnt[i] = 0;
}

__global__ void edge_slots(const int* __restrict__ ei, int* __restrict__ cnt,
                           int* __restrict__ slots) {
  int e = blockIdx.x * blockDim.x + threadIdx.x;
  if (e >= E_EDGES) return;
  int c = ei[e];                                  /* child order idx */
  int pos = atomicAdd(&cnt[c], 1);
  if (pos < MAXP) slots[c * MAXP + pos] = e;
}

__global__ void fill_f2p(const int* __restrict__ ei, const int* __restrict__ cnt,
                         const int* __restrict__ slots, float* __restrict__ out) {
  int c = blockIdx.x * blockDim.x + threadIdx.x;
  if (c >= N_O) return;
  int m = cnt[c]; if (m > MAXP) m = MAXP;
  int loc[MAXP];
  for (int i = 0; i < m; ++i) loc[i] = slots[c * MAXP + i];
  /* stable rank = ascending original edge index */
  for (int i = 1; i < m; ++i) {
    int key = loc[i]; int jx = i - 1;
    while (jx >= 0 && loc[jx] > key) { loc[jx + 1] = loc[jx]; --jx; }
    loc[jx + 1] = key;
  }
  const int* parent = ei + E_EDGES;               /* edge_index[1] */
  for (int r = 0; r < m; ++r) {
    float p = (float)parent[loc[r]];
#pragma unroll
    for (int j = 0; j < NCOLS; ++j) {
      long long row = (long long)NCOLS * N_U + (long long)j * N_O + c;
      out[OFF_F2P + row * MAXP + r] = p;
    }
  }
}

extern "C" void kernel_launch(void* const* d_in, const int* in_sizes, int n_in,
                              void* d_out, int out_size, void* d_ws, size_t ws_size,
                              hipStream_t stream) {
  const float* users_num   = (const float*)d_in[0];
  const int*   users_cat   = (const int*)  d_in[1];
  const float* users_text  = (const float*)d_in[2];
  const float* orders_num  = (const float*)d_in[3];
  const int*   orders_cat  = (const int*)  d_in[4];
  const float* orders_text = (const float*)d_in[5];
  const int*   edge_index  = (const int*)  d_in[6];
  const float* table_emb   = (const float*)d_in[7];
  const float* u_num_w   = (const float*)d_in[8];
  const float* u_num_b   = (const float*)d_in[9];
  const float* u_cat_tab = (const float*)d_in[10];
  const float* u_txt_w   = (const float*)d_in[11];
  const float* u_txt_b   = (const float*)d_in[12];
  const float* u_col     = (const float*)d_in[13];
  const float* o_num_w   = (const float*)d_in[14];
  const float* o_num_b   = (const float*)d_in[15];
  const float* o_cat_tab = (const float*)d_in[16];
  const float* o_txt_w   = (const float*)d_in[17];
  const float* o_txt_b   = (const float*)d_in[18];
  const float* o_col     = (const float*)d_in[19];
  float* out = (float*)d_out;

  /* numeric + categorical token columns (j = 0..5) */
  tok_numcat<<<N_U * (NNUM + NCAT), CDIM, 0, stream>>>(
      users_num, users_cat, u_cat_tab, u_num_w, u_num_b, u_col, table_emb,
      out, N_U, 0ll);
  tok_numcat<<<N_O * (NNUM + NCAT), CDIM, 0, stream>>>(
      orders_num, orders_cat, o_cat_tab, o_num_w, o_num_b, o_col, table_emb + CDIM,
      out, N_O, (long long)NCOLS * N_U);

  /* text token column (j = 6) via bf16 WMMA GEMM */
  dim3 gu((N_U + MTILE - 1) / MTILE, 2);
  dim3 go((N_O + MTILE - 1) / MTILE, 2);
  text_gemm_wmma<<<gu, 256, 0, stream>>>(
      users_text, u_txt_w, u_txt_b, u_col + 6 * CDIM, table_emb,
      out + (size_t)6 * N_U * CDIM, N_U);
  text_gemm_wmma<<<go, 256, 0, stream>>>(
      orders_text, o_txt_w, o_txt_b, o_col + 6 * CDIM, table_emb + CDIM,
      out + ((size_t)NCOLS * N_U + (size_t)6 * N_O) * CDIM, N_O);

  /* index arrays + num_nodes */
  idx_kernel<<<(NTOK + 255) / 256, 256, 0, stream>>>(out);

  /* neighbor table */
  long long nf2p = (long long)NTOK * MAXP;
  f2p_init<<<(int)((nf2p + 255) / 256), 256, 0, stream>>>(out);
  int* cnt   = (int*)d_ws;
  int* slots = cnt + N_O;
  zero_cnt<<<(N_O + 255) / 256, 256, 0, stream>>>(cnt);
  edge_slots<<<(E_EDGES + 255) / 256, 256, 0, stream>>>(edge_index, cnt, slots);
  fill_f2p<<<(N_O + 255) / 256, 256, 0, stream>>>(edge_index, cnt, slots, out);
}
